// MultiHeadedAttention_8761733283866
// MI455X (gfx1250) — matrix operand order repaired and checked by reading, not compiled, not run on hardware
//
#include <hip/hip_runtime.h>
#include <hip/hip_bf16.h>

// MI455X / gfx1250: wave32, WMMA bf16 16x16x32 with f32 accumulation.
// Multi-head attention B=2,S=2048,D=1024,H=16,DK=64.
// Compute-bound (~69 GFLOP vs ~120MB traffic @ 23.3TB/s) -> WMMA everywhere,
// bf16 operands staged through LDS, fp32 accumulation, fp32 I/O at the edges.
//
// Round-3 polish: fragment loads grouped ahead of each WMMA burst (so the
// scheduler can retire DS counters incrementally instead of s_wait_dscnt 0
// before every WMMA), and double-buffered LDS tiles in both kernels (one
// barrier per K-step, global->register fetch of tile i+1 overlapped with
// compute on tile i).

typedef __attribute__((ext_vector_type(16))) __bf16    v16bf;
typedef __attribute__((ext_vector_type(8)))  float     v8f;
typedef __attribute__((ext_vector_type(4)))  unsigned  u32x4;
typedef __attribute__((ext_vector_type(4)))  float     f32x4;
typedef __attribute__((ext_vector_type(4)))  int       i32x4;
typedef __attribute__((ext_vector_type(2)))  unsigned  u32x2;

union FragBF { u32x4 u[2]; v16bf v; };   // 16 bf16 = one WMMA A/B operand (8 VGPRs)

__device__ __forceinline__ unsigned short f2bf(float f) {
  union { float f; unsigned u; } x{f};
  unsigned r = x.u + 0x7fffu + ((x.u >> 16) & 1u);   // round-to-nearest-even
  return (unsigned short)(r >> 16);
}
__device__ __forceinline__ u32x2 pack4bf(float a, float b, float c, float d) {
  u32x2 r;
  r.x = (unsigned)f2bf(a) | ((unsigned)f2bf(b) << 16);
  r.y = (unsigned)f2bf(c) | ((unsigned)f2bf(d) << 16);
  return r;
}

// ---------------------------------------------------------------------------
// GEMM: C[4096,1024] = A[4096,1024] * W[1024,1024]^T + bias
// Block tile 128(M) x 64(N), K-step 32, 8 waves (wave w owns rows 16w..16w+15).
// Double-buffered LDS tiles; A_BF16/OUT_F32 select the projection vs final
// output-projection variant.
// ---------------------------------------------------------------------------
template <bool A_BF16, bool OUT_F32>
__global__ __launch_bounds__(256)
void gemm_xwt(const void* __restrict__ Aptr, const float* __restrict__ W,
              const float* __restrict__ bias, void* __restrict__ Out) {
  constexpr int Kdim = 1024, Ndim = 1024, NT = Kdim / 32;
  constexpr int SA = 40, SB = 40;                       // 80B rows (16B multiple)
  __shared__ __attribute__((aligned(16))) unsigned short As[2][128 * SA];
  __shared__ __attribute__((aligned(16))) unsigned short Bs[2][64 * SB];

  const int tid  = threadIdx.x;
  const int wave = tid >> 5, lane = tid & 31;
  const int half = lane >> 4, lh = lane & 15;
  const int m0 = blockIdx.y * 128;
  const int n0 = blockIdx.x * 64;

  // staged tile registers (global -> regs -> LDS)
  u32x2 aR[4];            // fp32-A path: 4x 4 bf16-packed values
  u32x4 aR16[2];          // bf16-A path: 2x 16B chunks
  u32x2 bR[2];            // W tile: 2x 4 bf16-packed values

  auto fetch = [&](int k0) {
    if constexpr (A_BF16) {
      const unsigned short* A16 = (const unsigned short*)Aptr;
#pragma unroll
      for (int i = 0; i < 2; ++i) {                      // 512 16B chunks / 256 thr
        int ch = tid * 2 + i;
        int row = ch >> 2, part = ch & 3;
        aR16[i] = *(const u32x4*)(A16 + (size_t)(m0 + row) * Kdim + k0 + part * 8);
      }
    } else {
      const float* A32 = (const float*)Aptr;
#pragma unroll
      for (int i = 0; i < 4; ++i) {                      // 1024 float4 / 256 thr
        int ch = tid * 4 + i;
        int row = ch >> 3, kq = ch & 7;
        f32x4 f = *(const f32x4*)(A32 + (size_t)(m0 + row) * Kdim + k0 + kq * 4);
        aR[i] = pack4bf(f.x, f.y, f.z, f.w);
      }
    }
#pragma unroll
    for (int i = 0; i < 2; ++i) {                        // W tile: 64 x 32 fp32
      int ch = tid * 2 + i;
      int n = ch >> 3, kq = ch & 7;
      f32x4 f = *(const f32x4*)(W + (size_t)(n0 + n) * Kdim + k0 + kq * 4);
      bR[i] = pack4bf(f.x, f.y, f.z, f.w);
    }
  };
  auto stash = [&](int buf) {
    unsigned short* As_ = As[buf];
    unsigned short* Bs_ = Bs[buf];
    if constexpr (A_BF16) {
#pragma unroll
      for (int i = 0; i < 2; ++i) {
        int ch = tid * 2 + i;
        int row = ch >> 2, part = ch & 3;
        *(u32x4*)(As_ + row * SA + part * 8) = aR16[i];
      }
    } else {
#pragma unroll
      for (int i = 0; i < 4; ++i) {
        int ch = tid * 4 + i;
        int row = ch >> 3, kq = ch & 7;
        *(u32x2*)(As_ + row * SA + kq * 4) = aR[i];
      }
    }
#pragma unroll
    for (int i = 0; i < 2; ++i) {
      int ch = tid * 2 + i;
      int n = ch >> 3, kq = ch & 7;
      *(u32x2*)(Bs_ + n * SB + kq * 4) = bR[i];
    }
  };

  const v8f zero = {};
  v8f acc[4];
#pragma unroll
  for (int t = 0; t < 4; ++t) acc[t] = zero;

  fetch(0);
  stash(0);
  int cur = 0;
  for (int it = 0; it < NT; ++it) {
    __syncthreads();
    if (it + 1 < NT) fetch((it + 1) * 32);               // overlap with compute

    const unsigned short* As_ = As[cur];
    const unsigned short* Bs_ = Bs[cur];
    // A fragment: 16x32, lane lh = row; half 0: K{0..7,16..23}, half 1: +8
    FragBF a;
    const unsigned short* ap = As_ + (wave * 16 + lh) * SA + half * 8;
    a.u[0] = *(const u32x4*)(ap);
    a.u[1] = *(const u32x4*)(ap + 16);
    // all 4 B fragments first, then the WMMA burst
    FragBF bf[4];
#pragma unroll
    for (int t = 0; t < 4; ++t) {
      const unsigned short* bp = Bs_ + (t * 16 + lh) * SB + half * 8;
      bf[t].u[0] = *(const u32x4*)(bp);
      bf[t].u[1] = *(const u32x4*)(bp + 16);
    }
#pragma unroll
    for (int t = 0; t < 4; ++t)
      acc[t] = __builtin_amdgcn_wmma_f32_16x16x32_bf16(
          false, a.v, false, bf[t].v, (short)0, acc[t], false, false);

    if (it + 1 < NT) stash(cur ^ 1);
    cur ^= 1;
  }

  // C layout: VGPR r -> row r (lanes 0-15) / r+8 (lanes 16-31), col = lane&15
#pragma unroll
  for (int t = 0; t < 4; ++t) {
    int colg = n0 + t * 16 + lh;
    float bv = bias[colg];
#pragma unroll
    for (int r = 0; r < 8; ++r) {
      int rowg = m0 + wave * 16 + r + half * 8;
      float v = acc[t][r] + bv;
      if constexpr (OUT_F32) {
        ((float*)Out)[(size_t)rowg * Ndim + colg] = v;
      } else {
        int bb = rowg >> 11, s = rowg & 2047;            // row = b*2048 + s
        int hh = colg >> 6, dk = colg & 63;              // col = h*64 + dk
        ((unsigned short*)Out)[(((size_t)(bb * 16 + hh) * 2048) + s) * 64 + dk] =
            f2bf(v);
      }
    }
  }
}

// ---------------------------------------------------------------------------
// Flash attention: per (b,h), 128-row query tile per block (16 rows per wave),
// streaming 32-key tiles (double-buffered). bf16 WMMA for QK^T and PV, online
// softmax in fp32, lane-local in A-layout after the LDS score transpose.
// ---------------------------------------------------------------------------
__global__ __launch_bounds__(256)
void attn_kernel(const unsigned short* __restrict__ qb,
                 const unsigned short* __restrict__ kb,
                 const unsigned short* __restrict__ vb,
                 const int* __restrict__ mask,
                 unsigned short* __restrict__ ob) {
  constexpr int S = 2048, NT = S / 32;
  constexpr int SK = 72, SV = 40, SSC = 36;              // 144B / 80B / 144B rows
  __shared__ __attribute__((aligned(16))) unsigned short Ks[2][32 * SK];  // [key][dk]
  __shared__ __attribute__((aligned(16))) unsigned short Vt[2][64 * SV];  // [dk][key]
  __shared__ __attribute__((aligned(16))) int Ms[2][128 * 32];            // mask tile
  __shared__ __attribute__((aligned(16))) float Sc[8][16 * SSC];          // fp32 scores

  const int tid = threadIdx.x, wave = tid >> 5, lane = tid & 31;
  const int half = lane >> 4, lh = lane & 15;
  const int bh = blockIdx.y;
  const int b = bh >> 4, h = bh & 15;
  const int q0 = blockIdx.x * 128;

  const unsigned short* qh = qb + (size_t)bh * S * 64;
  const unsigned short* kh = kb + (size_t)bh * S * 64;
  const unsigned short* vh = vb + (size_t)bh * S * 64;

  // staged tile registers
  u32x4 kR, vR;
  i32x4 mR[4];
  auto fetchT = [&](int t0) {
    int key = tid >> 3, part = tid & 7;
    kR = *(const u32x4*)(kh + (size_t)(t0 + key) * 64 + part * 8);
    vR = *(const u32x4*)(vh + (size_t)(t0 + key) * 64 + part * 8);
#pragma unroll
    for (int i = 0; i < 4; ++i) {
      int ch = tid * 4 + i;
      int row = ch >> 3, cq = ch & 7;
      mR[i] = *(const i32x4*)(mask + (size_t)b * S * S + (size_t)(q0 + row) * S +
                              t0 + cq * 4);
    }
  };
  auto stashT = [&](int buf) {
    int key = tid >> 3, part = tid & 7;
    *(u32x4*)(Ks[buf] + key * SK + part * 8) = kR;       // [key][dk]
    const unsigned short* pv = (const unsigned short*)&vR;
#pragma unroll
    for (int j = 0; j < 8; ++j) Vt[buf][(part * 8 + j) * SV + key] = pv[j];  // [dk][key]
#pragma unroll
    for (int i = 0; i < 4; ++i) {
      int ch = tid * 4 + i;
      int row = ch >> 3, cq = ch & 7;
      *(i32x4*)(Ms[buf] + row * 32 + cq * 4) = mR[i];
    }
  };

  // Q fragments (16 rows x dk64): A-layout, held in registers for whole pass
  FragBF qf[2];
  {
    const unsigned short* qp = qh + (size_t)(q0 + wave * 16 + lh) * 64;
    qf[0].u[0] = *(const u32x4*)(qp + half * 8);
    qf[0].u[1] = *(const u32x4*)(qp + 16 + half * 8);
    qf[1].u[0] = *(const u32x4*)(qp + 32 + half * 8);
    qf[1].u[1] = *(const u32x4*)(qp + 48 + half * 8);
  }

  const v8f zero = {};
  v8f o[4];
#pragma unroll
  for (int j = 0; j < 4; ++j) o[j] = zero;
  float mrun = -3.0e38f, lrun = 0.f;                     // row = wave*16 + lh

  const float scale = 0.125f;                            // 1/sqrt(64)
  const float NEGV = -1e9f;

  fetchT(0);
  stashT(0);
  int cur = 0;
  for (int it = 0; it < NT; ++it) {
    __syncthreads();
    if (it + 1 < NT) fetchT((it + 1) * 32);              // overlap with compute

    // scores: Q(16x64) * K^T -> two 16x16 accumulators (keys 0-15, 16-31).
    // Load all 4 B fragments, then the WMMA burst.
    FragBF kb0[2], kb1[2];
#pragma unroll
    for (int ks = 0; ks < 2; ++ks) {
      const unsigned short* kp0 = Ks[cur] + lh * SK + ks * 32 + half * 8;
      const unsigned short* kp1 = Ks[cur] + (16 + lh) * SK + ks * 32 + half * 8;
      kb0[ks].u[0] = *(const u32x4*)kp0; kb0[ks].u[1] = *(const u32x4*)(kp0 + 16);
      kb1[ks].u[0] = *(const u32x4*)kp1; kb1[ks].u[1] = *(const u32x4*)(kp1 + 16);
    }
    v8f s0 = zero, s1 = zero;
    s0 = __builtin_amdgcn_wmma_f32_16x16x32_bf16(false, qf[0].v, false, kb0[0].v,
                                                 (short)0, s0, false, false);
    s1 = __builtin_amdgcn_wmma_f32_16x16x32_bf16(false, qf[0].v, false, kb1[0].v,
                                                 (short)0, s1, false, false);
    s0 = __builtin_amdgcn_wmma_f32_16x16x32_bf16(false, qf[1].v, false, kb0[1].v,
                                                 (short)0, s0, false, false);
    s1 = __builtin_amdgcn_wmma_f32_16x16x32_bf16(false, qf[1].v, false, kb1[1].v,
                                                 (short)0, s1, false, false);

    // C-layout store of raw scores to per-wave LDS scratch (the transpose)
#pragma unroll
    for (int r = 0; r < 8; ++r) {
      int mrow = r + half * 8;
      Sc[wave][mrow * SSC + lh]      = s0[r];
      Sc[wave][mrow * SSC + 16 + lh] = s1[r];
    }
    asm volatile("s_wait_dscnt 0" ::: "memory");         // same-wave store->load

    // A-layout reload: this lane = row lh, cols {h8..h8+7, 16+h8..16+h8+7}
    const float* sp = &Sc[wave][lh * SSC + half * 8];
    f32x4 c0 = *(const f32x4*)(sp);
    f32x4 c1 = *(const f32x4*)(sp + 4);
    f32x4 c2 = *(const f32x4*)(sp + 16);
    f32x4 c3 = *(const f32x4*)(sp + 20);
    const int* mp = &Ms[cur][(wave * 16 + lh) * 32 + half * 8];
    i32x4 mv0 = *(const i32x4*)(mp);
    i32x4 mv1 = *(const i32x4*)(mp + 4);
    i32x4 mv2 = *(const i32x4*)(mp + 16);
    i32x4 mv3 = *(const i32x4*)(mp + 20);

    float sv[16];
#pragma unroll
    for (int e = 0; e < 4; ++e) {
      sv[e]      = (mv0[e] == 0) ? NEGV : c0[e] * scale;
      sv[4 + e]  = (mv1[e] == 0) ? NEGV : c1[e] * scale;
      sv[8 + e]  = (mv2[e] == 0) ? NEGV : c2[e] * scale;
      sv[12 + e] = (mv3[e] == 0) ? NEGV : c3[e] * scale;
    }

    // lane-local row reduction + one half-swap shuffle (lane <-> lane+16)
    float tm = sv[0];
#pragma unroll
    for (int e = 1; e < 16; ++e) tm = fmaxf(tm, sv[e]);
    tm = fmaxf(tm, __shfl_xor(tm, 16, 32));
    float mn = fmaxf(mrun, tm);
    float al = __expf(mrun - mn);
    mrun = mn;

    float p[16], rs = 0.f;
#pragma unroll
    for (int e = 0; e < 16; ++e) { p[e] = __expf(sv[e] - mn); rs += p[e]; }
    rs += __shfl_xor(rs, 16, 32);
    lrun = lrun * al + rs;

    // P fragment built directly in registers (already in A-layout order)
    FragBF pf;
#pragma unroll
    for (int e = 0; e < 4; ++e) {
      pf.u[0][e] = (unsigned)f2bf(p[2 * e]) | ((unsigned)f2bf(p[2 * e + 1]) << 16);
      pf.u[1][e] = (unsigned)f2bf(p[8 + 2 * e]) |
                   ((unsigned)f2bf(p[8 + 2 * e + 1]) << 16);
    }

    // rescale O: alpha lives row-per-lane, O needs it row-per-VGPR -> broadcast
#pragma unroll
    for (int r = 0; r < 8; ++r) {
      float alr = __shfl(al, r + half * 8, 32);
#pragma unroll
      for (int j = 0; j < 4; ++j) o[j][r] *= alr;
    }

    // O += P(16x32) * V(32x64): all 4 V fragments, then the WMMA burst
    FragBF vf[4];
#pragma unroll
    for (int j = 0; j < 4; ++j) {
      const unsigned short* vp = Vt[cur] + (j * 16 + lh) * SV + half * 8;
      vf[j].u[0] = *(const u32x4*)vp;
      vf[j].u[1] = *(const u32x4*)(vp + 16);
    }
#pragma unroll
    for (int j = 0; j < 4; ++j)
      o[j] = __builtin_amdgcn_wmma_f32_16x16x32_bf16(false, pf.v, false, vf[j].v,
                                                     (short)0, o[j], false, false);

    if (it + 1 < NT) stashT(cur ^ 1);
    cur ^= 1;
  }

  // normalize and write bf16 [B,S,D] (D col = h*64 + dk)
  float inv = 1.0f / lrun;
#pragma unroll
  for (int r = 0; r < 8; ++r) {
    float invr = __shfl(inv, r + half * 8, 32);
    int rowg = q0 + wave * 16 + r + half * 8;
#pragma unroll
    for (int j = 0; j < 4; ++j) {
      int colg = h * 64 + j * 16 + lh;
      ob[((size_t)b * S + rowg) * 1024 + colg] = f2bf(o[j][r] * invr);
    }
  }
}

// ---------------------------------------------------------------------------
extern "C" void kernel_launch(void* const* d_in, const int* in_sizes, int n_in,
                              void* d_out, int out_size, void* d_ws,
                              size_t ws_size, hipStream_t stream) {
  const float* query = (const float*)d_in[0];
  const float* key_  = (const float*)d_in[1];
  const float* value = (const float*)d_in[2];
  const int*   mask  = (const int*)d_in[3];
  const float* Wq = (const float*)d_in[4];
  const float* bq = (const float*)d_in[5];
  const float* Wk = (const float*)d_in[6];
  const float* bk = (const float*)d_in[7];
  const float* Wv = (const float*)d_in[8];
  const float* bv = (const float*)d_in[9];
  const float* Wo = (const float*)d_in[10];
  const float* bo = (const float*)d_in[11];

  const size_t HEADBUF = (size_t)2 * 16 * 2048 * 64;     // 4M bf16 elems each
  unsigned short* qb = (unsigned short*)d_ws;
  unsigned short* kb = qb + HEADBUF;
  unsigned short* vb = kb + HEADBUF;
  unsigned short* ob = vb + HEADBUF;                     // 32 MB total workspace

  dim3 blk(256);
  dim3 gg(16, 32);                                       // N tiles x M tiles
  gemm_xwt<false, false><<<gg, blk, 0, stream>>>(query, Wq, bq, qb);
  gemm_xwt<false, false><<<gg, blk, 0, stream>>>(key_,  Wk, bk, kb);
  gemm_xwt<false, false><<<gg, blk, 0, stream>>>(value, Wv, bv, vb);
  attn_kernel<<<dim3(16, 32), blk, 0, stream>>>(qb, kb, vb, mask, ob);
  gemm_xwt<true, true><<<gg, blk, 0, stream>>>(ob, Wo, bo, d_out);
}
